// AttentionFusion_10711648436471
// MI455X (gfx1250) — matrix operand order repaired and checked by reading, not compiled, not run on hardware
//
#include <hip/hip_runtime.h>
#include <hip/hip_bf16.h>
#include <stdint.h>

// Problem constants (match reference)
#define BDIM 32768
#define MDIM 4
#define DDIM 1024
#define HDIM 8
#define DH   128
#define EPSLN 1e-5f

// Tiling
#define BT    16              // batch elements per workgroup
#define ROWS  64              // BT * MDIM rows per workgroup
#define AST   1032            // bf16 row stride for x/ctx LDS tiles (stride%64dw==4 -> conflict-free b128)
#define SST   136             // bf16 row stride for per-head q/k/v stage (272B, 16B aligned)
#define KT    32              // 1024 / 32 k-steps

typedef __bf16   v16bf __attribute__((ext_vector_type(16)));
typedef float    v8f   __attribute__((ext_vector_type(8)));
typedef uint32_t v4u   __attribute__((ext_vector_type(4)));
typedef uint32_t v2u   __attribute__((ext_vector_type(2)));

struct Frag { v4u lo, hi; };

static __device__ __forceinline__ uint16_t f2bf(float f) {
  uint32_t u = __builtin_bit_cast(uint32_t, f);
  uint32_t r = (u + 0x7FFFu + ((u >> 16) & 1u)) >> 16;   // RNE
  return (uint16_t)r;
}
static __device__ __forceinline__ uint32_t pack2(float a, float b) {
  return (uint32_t)f2bf(a) | ((uint32_t)f2bf(b) << 16);
}
static __device__ __forceinline__ float bflo(uint32_t u) {
  return __builtin_bit_cast(float, u << 16);
}
static __device__ __forceinline__ float bfhi(uint32_t u) {
  return __builtin_bit_cast(float, u & 0xffff0000u);
}

// A-matrix fragment (16x32 bf16, ISA 7.12.2): lane m=lane&15, half=lane>>4.
// lanes<16: K 0..7 @ +0B, K 16..23 @ +32B ; lanes>=16: K 8..15 @ +16B, K 24..31 @ +48B.
static __device__ __forceinline__ Frag load_a_lds(const uint16_t* base, int row0,
                                                  int kt, int lane, int rstride) {
  int m = lane & 15, hf = lane >> 4;
  const char* p = (const char*)(base + (row0 + m) * rstride + kt * 32) + hf * 16;
  Frag f;
  f.lo = *(const v4u*)p;
  f.hi = *(const v4u*)(p + 32);
  return f;
}

// B-matrix fragment (32x16 bf16): lane n=lane&15, same K order as A
// (K 0..7,16..23 for lanes<16, K 8..15,24..31 for lanes>=16) -> two b128 from row-major W[n][k].
static __device__ __forceinline__ Frag load_b_glb(const uint16_t* w, int nbase,
                                                  int kt, int lane) {
  int n = nbase + (lane & 15), hf = lane >> 4;
  const uint16_t* p = w + n * DDIM + kt * 32 + hf * 8;
  Frag f;
  f.lo = *(const v4u*)p;
  f.hi = *(const v4u*)(p + 16);
  return f;
}

static __device__ __forceinline__ v8f wmma_bf16(const Frag& a, const Frag& b, v8f c) {
  v16bf av = __builtin_bit_cast(v16bf, a);
  v16bf bv = __builtin_bit_cast(v16bf, b);
  return __builtin_amdgcn_wmma_f32_16x16x32_bf16(false, av, false, bv,
                                                 (short)0, c, false, false);
}

static __device__ __forceinline__ v8f zero8() {
  v8f z = {0.f, 0.f, 0.f, 0.f, 0.f, 0.f, 0.f, 0.f};
  return z;
}

// Software-pipelined K loop: per k-step, issue all NJ B-fragment loads (global,
// in-order -> partial s_wait_loadcnt before each wmma) plus the NEXT k-step's A
// fragment (DScnt, consumed next iteration), then run the wmma chain.
template <int NJ>
static __device__ __forceinline__ void gemm_kloop(const uint16_t* aLDS, int row0,
                                                  const uint16_t* w,
                                                  const int (&nb)[NJ],
                                                  int lane, v8f (&acc)[NJ]) {
  Frag a = load_a_lds(aLDS, row0, 0, lane, AST);
  #pragma unroll 1
  for (int kt = 0; kt < KT; ++kt) {
    Frag bfr[NJ];
    #pragma unroll
    for (int j = 0; j < NJ; ++j) bfr[j] = load_b_glb(w, nb[j], kt, lane);
    int ktn = (kt + 1 < KT) ? kt + 1 : kt;
    Frag an = load_a_lds(aLDS, row0, ktn, lane, AST);
    #pragma unroll
    for (int j = 0; j < NJ; ++j) acc[j] = wmma_bf16(a, bfr[j], acc[j]);
    a = an;
  }
}

// ---------------- weight f32 -> bf16 pre-pass ----------------
__global__ void cvt_f32_bf16(const float* __restrict__ s, uint16_t* __restrict__ d, int n) {
  int i = blockIdx.x * blockDim.x + threadIdx.x;
  if (i < n) d[i] = f2bf(s[i]);
}

// ---------------- fused attention kernel ----------------
__global__ __launch_bounds__(256)
void attn_fuse_kernel(const float* __restrict__ x,
                      const float* __restrict__ in_b,
                      const float* __restrict__ out_b,
                      const float* __restrict__ g_b1,
                      const float* __restrict__ g_w2,
                      const float* __restrict__ g_b2,
                      const float* __restrict__ lnw,
                      const float* __restrict__ lnb,
                      const uint16_t* __restrict__ wqkv,   // [3072][1024] bf16
                      const uint16_t* __restrict__ wout,   // [1024][1024] bf16
                      const uint16_t* __restrict__ w1,     // [512][1024] bf16
                      float* __restrict__ out) {
  extern __shared__ char smem[];
  uint16_t* xA    = (uint16_t*)smem;                       // 64 x 1032 bf16 (132096 B); later fused[16][1024] f32
  uint16_t* ctx   = (uint16_t*)(smem + ROWS * AST * 2);    // 64 x 1032 bf16 (132096 B)
  uint16_t* stage = (uint16_t*)(smem + 2 * ROWS * AST * 2);// 3 x 64 x 136 bf16 (52224 B); later gsum/g

  const int tid  = threadIdx.x;
  const int wave = tid >> 5;
  const int lane = tid & 31;
  const int rb   = wave & 3;        // row-block (16 rows)
  const int cg   = wave >> 2;       // column group

  // ---- Phase 1: stage x tile [64 x 1024] f32 -> bf16 LDS ----
  {
    int row = tid >> 2;
    int cb  = (tid & 3) * 256;
    const float* xg = x + ((size_t)blockIdx.x * ROWS + row) * DDIM;
    uint16_t* xr = xA + row * AST;
    #pragma unroll 4
    for (int i = 0; i < 64; ++i) {
      int col = cb + i * 4;
      float4 v = *(const float4*)(xg + col);
      v2u p;
      p.x = pack2(v.x, v.y);
      p.y = pack2(v.z, v.w);
      *(v2u*)(xr + col) = p;
    }
  }
  __syncthreads();

  // ---- Phase 2: per-head QKV GEMM (WMMA) + attention -> ctx ----
  for (int h = 0; h < HDIM; ++h) {
    // 24 col-blocks (8 q, 8 k, 8 v) x 4 row-blocks; wave does rb x 12 col-blocks.
    for (int pass = 0; pass < 2; ++pass) {
      v8f acc[6];
      #pragma unroll
      for (int j = 0; j < 6; ++j) acc[j] = zero8();
      int cbs = cg * 12 + pass * 6;
      int nb[6];
      #pragma unroll
      for (int j = 0; j < 6; ++j) {
        int cb = cbs + j;
        nb[j] = (cb >> 3) * DDIM + h * DH + (cb & 7) * 16;
      }
      gemm_kloop<6>(xA, rb * 16, wqkv, nb, lane, acc);
      // epilogue: +bias, store bf16 into per-head q/k/v stage
      #pragma unroll
      for (int j = 0; j < 6; ++j) {
        int cb = cbs + j, seg = cb >> 3, dh0 = (cb & 7) * 16;
        float bias = in_b[nb[j] + (lane & 15)];
        uint16_t* sseg = stage + seg * (ROWS * SST);
        int hf = lane >> 4, col = dh0 + (lane & 15);
        #pragma unroll
        for (int i = 0; i < 8; ++i) {
          int r = rb * 16 + hf * 8 + i;     // C-frag row map (ISA 7.12.2)
          sseg[r * SST + col] = f2bf(acc[j][i] + bias);
        }
      }
    }
    __syncthreads();

    // attention for head h: thread t -> (batch bb, qm, km); one 4x4 score each
    {
      int bb = tid >> 4;
      int jj = tid & 15;
      int qm = jj >> 2, km = jj & 3;
      const uint16_t* qs = stage;
      const uint16_t* ks = stage + ROWS * SST;
      const uint16_t* vs = stage + 2 * ROWS * SST;
      int rq = bb * 4 + qm, rk = bb * 4 + km;
      float s = 0.f;
      #pragma unroll
      for (int i = 0; i < 16; ++i) {
        v4u qa = *(const v4u*)(qs + rq * SST + i * 8);
        v4u ka = *(const v4u*)(ks + rk * SST + i * 8);
        s += bflo(qa.x) * bflo(ka.x) + bfhi(qa.x) * bfhi(ka.x);
        s += bflo(qa.y) * bflo(ka.y) + bfhi(qa.y) * bfhi(ka.y);
        s += bflo(qa.z) * bflo(ka.z) + bfhi(qa.z) * bfhi(ka.z);
        s += bflo(qa.w) * bflo(ka.w) + bfhi(qa.w) * bfhi(ka.w);
      }
      s *= 0.08838834764831845f;            // 1/sqrt(128)
      float mx = s;
      mx = fmaxf(mx, __shfl_xor(mx, 1));
      mx = fmaxf(mx, __shfl_xor(mx, 2));    // softmax across km (lane bits 0..1)
      float e = __expf(s - mx);
      float den = e;
      den += __shfl_xor(den, 1);
      den += __shfl_xor(den, 2);
      float attn = e / den;
      float aw[4];
      #pragma unroll
      for (int mi = 0; mi < 4; ++mi) aw[mi] = __shfl(attn, (lane & ~3) | mi);
      // ctx[rq, h*128 + km*32 .. +31] = sum_mi aw[mi] * v[bb*4+mi, :]
      int colh = h * DH + km * 32;
      #pragma unroll
      for (int c = 0; c < 4; ++c) {
        float o[8] = {0.f, 0.f, 0.f, 0.f, 0.f, 0.f, 0.f, 0.f};
        #pragma unroll
        for (int mi = 0; mi < 4; ++mi) {
          v4u vv = *(const v4u*)(vs + (bb * 4 + mi) * SST + km * 32 + c * 8);
          float am = aw[mi];
          o[0] += am * bflo(vv.x); o[1] += am * bfhi(vv.x);
          o[2] += am * bflo(vv.y); o[3] += am * bfhi(vv.y);
          o[4] += am * bflo(vv.z); o[5] += am * bfhi(vv.z);
          o[6] += am * bflo(vv.w); o[7] += am * bfhi(vv.w);
        }
        v4u pk;
        pk.x = pack2(o[0], o[1]); pk.y = pack2(o[2], o[3]);
        pk.z = pack2(o[4], o[5]); pk.w = pack2(o[6], o[7]);
        *(v4u*)(ctx + rq * AST + colh + c * 8) = pk;
      }
    }
    __syncthreads();
  }

  // ---- Phase 3: gate MLP (WMMA) -> g[64] ----
  float* gsum = (float*)stage;        // stage region is free now
  float* gsh  = (float*)stage + 64;
  if (tid < 64) gsum[tid] = 0.f;
  __syncthreads();
  for (int pass = 0; pass < 4; ++pass) {
    v8f acc[4];
    #pragma unroll
    for (int j = 0; j < 4; ++j) acc[j] = zero8();
    int cbs = cg * 16 + pass * 4;     // 32 col-blocks of W1 total
    int nb[4];
    #pragma unroll
    for (int j = 0; j < 4; ++j) nb[j] = (cbs + j) * 16;
    gemm_kloop<4>(xA, rb * 16, w1, nb, lane, acc);
    #pragma unroll
    for (int j = 0; j < 4; ++j) {
      int n = nb[j] + (lane & 15);
      float b1 = g_b1[n];
      float w2 = g_w2[n];
      int hf = lane >> 4;
      #pragma unroll
      for (int i = 0; i < 8; ++i) {
        float hv = acc[j][i] + b1;
        hv = hv > 0.f ? hv : 0.f;                     // relu
        float contrib = hv * w2;
        #pragma unroll
        for (int o = 1; o < 16; o <<= 1) contrib += __shfl_xor(contrib, o);
        if ((lane & 15) == 0)
          atomicAdd(&gsum[rb * 16 + hf * 8 + i], contrib);
      }
    }
  }
  __syncthreads();
  if (tid < 64) {
    float z = gsum[tid] + g_b2[0];
    gsh[tid] = 1.f / (1.f + __expf(-z));              // sigmoid
  }
  __syncthreads();

  // ---- Phase 4: out_proj (WMMA) + gated modality sum -> fused[16][1024] f32 ----
  float* fused = (float*)xA;          // overlay: x tile no longer needed
  for (int pass = 0; pass < 8; ++pass) {
    v8f acc[4];
    #pragma unroll
    for (int j = 0; j < 4; ++j) acc[j] = zero8();
    int cbs = cg * 32 + pass * 4;     // 64 col-blocks total
    int nb[4];
    #pragma unroll
    for (int j = 0; j < 4; ++j) nb[j] = (cbs + j) * 16;
    gemm_kloop<4>(ctx, rb * 16, wout, nb, lane, acc);
    #pragma unroll
    for (int j = 0; j < 4; ++j) {
      int n = nb[j] + (lane & 15);
      float bias = out_b[n];
      int hf = lane >> 4;
      int ba = rb * 4 + hf * 2;       // rows rb*16+hf*8+0..3 -> batch ba ; +4..7 -> ba+1
      float fA = 0.f, fB = 0.f;
      #pragma unroll
      for (int i = 0; i < 4; ++i) fA += (acc[j][i] + bias) * gsh[rb * 16 + hf * 8 + i];
      #pragma unroll
      for (int i = 4; i < 8; ++i) fB += (acc[j][i] + bias) * gsh[rb * 16 + hf * 8 + i];
      fused[ba * DDIM + n]       = fA;
      fused[(ba + 1) * DDIM + n] = fB;
    }
  }
  __syncthreads();

  // ---- Phase 5: LayerNorm over D=1024, write output ----
  for (int rep = 0; rep < 2; ++rep) {
    int bb = wave * 2 + rep;
    const float* fr = fused + bb * DDIM;
    float s = 0.f, s2 = 0.f;
    #pragma unroll 4
    for (int i = 0; i < 32; ++i) {
      float v = fr[i * 32 + lane];
      s += v; s2 += v * v;
    }
    #pragma unroll
    for (int o = 1; o < 32; o <<= 1) { s += __shfl_xor(s, o); s2 += __shfl_xor(s2, o); }
    float mu   = s * (1.f / 1024.f);
    float var  = s2 * (1.f / 1024.f) - mu * mu;
    float rstd = rsqrtf(var + EPSLN);
    float* og = out + ((size_t)blockIdx.x * BT + bb) * DDIM;
    #pragma unroll 4
    for (int i = 0; i < 32; ++i) {
      int col = i * 32 + lane;
      og[col] = (fr[col] - mu) * rstd * lnw[col] + lnb[col];
    }
  }
}

extern "C" void kernel_launch(void* const* d_in, const int* in_sizes, int n_in,
                              void* d_out, int out_size, void* d_ws, size_t ws_size,
                              hipStream_t stream) {
  const float* x     = (const float*)d_in[0];
  const float* in_w  = (const float*)d_in[1];
  const float* in_b  = (const float*)d_in[2];
  const float* out_w = (const float*)d_in[3];
  const float* out_b = (const float*)d_in[4];
  const float* g_w1  = (const float*)d_in[5];
  const float* g_b1  = (const float*)d_in[6];
  const float* g_w2  = (const float*)d_in[7];
  const float* g_b2  = (const float*)d_in[8];
  const float* lnw   = (const float*)d_in[9];
  const float* lnb   = (const float*)d_in[10];

  uint16_t* ws   = (uint16_t*)d_ws;
  uint16_t* wqkv = ws;                         // 3*1024*1024 bf16
  uint16_t* wout = ws + 3 * 1024 * 1024;       // 1024*1024 bf16
  uint16_t* w1   = ws + 4 * 1024 * 1024;       // 512*1024 bf16

  const int nq = 3 * 1024 * 1024, no = 1024 * 1024, ng = 512 * 1024;
  cvt_f32_bf16<<<(nq + 255) / 256, 256, 0, stream>>>(in_w, wqkv, nq);
  cvt_f32_bf16<<<(no + 255) / 256, 256, 0, stream>>>(out_w, wout, no);
  cvt_f32_bf16<<<(ng + 255) / 256, 256, 0, stream>>>(g_w1, w1, ng);

  size_t smem = (size_t)(2 * ROWS * AST * 2 + 3 * ROWS * SST * 2);   // 316416 B
  attn_fuse_kernel<<<BDIM / BT, 256, smem, stream>>>(
      x, in_b, out_b, g_b1, g_w2, g_b2, lnw, lnb, wqkv, wout, w1, (float*)d_out);
}
